// BindingReadout_23270132810200
// MI455X (gfx1250) — compile-verified, matrix operand order checked by reading, not run on hardware
//
#include <hip/hip_runtime.h>
#include <hip/hip_bf16.h>

#define S_ 32
#define B_ 32
#define N_ 8192
#define D_ 256
#define M_ 8           // MAX_OBJECTS
#define CHUNKS 16
#define ROWS_PER_BLOCK (N_ / CHUNKS)   // 512

typedef __attribute__((ext_vector_type(2)))  float    v2f;
typedef __attribute__((ext_vector_type(8)))  float    v8f;
typedef __attribute__((ext_vector_type(16))) _Float16 v16h;

// workspace layout (in floats)
#define WS_SUMS 0
#define WS_CNT  (B_ * S_ * D_)                // 262144
#define WS_OBJS (WS_CNT + B_ * S_)            // 263168
#define WS_ZERO_N (WS_OBJS)                   // zero sums + counts each launch

#if defined(__has_builtin)
#if __has_builtin(__builtin_amdgcn_wmma_f32_16x16x4_f32)
#define HAVE_F32_WMMA 1
#endif
#endif

// ---------------------------------------------------------------------------
// Phase 0: zero the global accumulators (ws is NOT re-poisoned between calls)
// ---------------------------------------------------------------------------
__global__ __launch_bounds__(256) void zero_ws(float* __restrict__ ws) {
    int i = blockIdx.x * blockDim.x + threadIdx.x;
    int stride = gridDim.x * blockDim.x;
    for (; i < WS_ZERO_N; i += stride) ws[i] = 0.0f;
}

// ---------------------------------------------------------------------------
// Phase 1: segment sums + counts. One block = one 512-row chunk of one batch.
// LDS accumulator (S*D fp32 = 32KB) absorbs the scatter; one global atomic
// flush per block. Lane L owns columns [8L, 8L+8) -> 2x global_load_b128,
// fully coalesced 1KB per wave-row. This phase is the memory-bound bulk
// (256MB @ 23.3TB/s ~= 11us floor).
// ---------------------------------------------------------------------------
__global__ __launch_bounds__(256) void seg_accum(const float* __restrict__ features,
                                                 const int*   __restrict__ seg,
                                                 float*       __restrict__ ws) {
    __shared__ float acc[S_ * D_];
    __shared__ float cnt[S_];
    const int tid = threadIdx.x;
    for (int i = tid; i < S_ * D_; i += 256) acc[i] = 0.0f;
    if (tid < S_) cnt[tid] = 0.0f;
    __syncthreads();

    const int b     = blockIdx.x / CHUNKS;
    const int chunk = blockIdx.x % CHUNKS;
    const int wave  = tid >> 5;
    const int lane  = tid & 31;
    const float* fb = features + (size_t)b * N_ * D_;
    const int*   sb = seg + b * N_;

    for (int i = 0; i < ROWS_PER_BLOCK / 8; ++i) {
        const int n = chunk * ROWS_PER_BLOCK + i * 8 + wave;
        const int s = sb[n];
        const float4* row = (const float4*)(fb + (size_t)n * D_);
        float4 v0 = row[lane * 2 + 0];
        float4 v1 = row[lane * 2 + 1];
        float* a = acc + s * D_ + lane * 8;
        atomicAdd(a + 0, v0.x); atomicAdd(a + 1, v0.y);
        atomicAdd(a + 2, v0.z); atomicAdd(a + 3, v0.w);
        atomicAdd(a + 4, v1.x); atomicAdd(a + 5, v1.y);
        atomicAdd(a + 6, v1.z); atomicAdd(a + 7, v1.w);
        if (lane == 0) atomicAdd(&cnt[s], 1.0f);
    }
    __syncthreads();

    float* gsum = ws + WS_SUMS + (size_t)b * S_ * D_;
    for (int i = tid; i < S_ * D_; i += 256) atomicAdd(&gsum[i], acc[i]);
    if (tid < S_) atomicAdd(&ws[WS_CNT + b * S_ + tid], cnt[tid]);
}

// ---------------------------------------------------------------------------
// Phase 2a: stable top-8 selection (== argsort(-counts, stable)[:8]) and
// materialize objs[b*8+m][d] = sums[b][top[m]][d] * (cnt>0 ? 1/cnt : 0).
// ---------------------------------------------------------------------------
__global__ __launch_bounds__(256) void top_objs(float* __restrict__ ws) {
    __shared__ int   topi[M_];
    __shared__ float scale[M_];
    const int b = blockIdx.x, tid = threadIdx.x;
    if (tid == 0) {
        const float* c = ws + WS_CNT + b * S_;
        unsigned used = 0u;
        for (int m = 0; m < M_; ++m) {
            int best = 0; float bc = -1.0f;
            for (int s = 0; s < S_; ++s) {
                if (!((used >> s) & 1u) && c[s] > bc) { bc = c[s]; best = s; }
            }
            used |= (1u << best);
            topi[m]  = best;
            scale[m] = (bc > 0.0f) ? (1.0f / bc) : 0.0f;
        }
    }
    __syncthreads();
    const float* gsum = ws + WS_SUMS + (size_t)b * S_ * D_;
    float*       objs = ws + WS_OBJS + (size_t)b * M_ * D_;
    for (int m = 0; m < M_; ++m)
        objs[m * D_ + tid] = gsum[topi[m] * D_ + tid] * scale[m];
}

// ---------------------------------------------------------------------------
// Phase 2b: proj = objs (256x256) * W^T (256x256) + b, then LayerNorm.
// Block = 512 threads = 16 waves; wave w owns the 16x16 tile at columns
// [16w, 16w+16) of a 16-row strip -> the whole 16x256 strip lives in one
// block so LayerNorm can reduce over the full row in LDS.
// Primary path: exact fp32 V_WMMA_F32_16X16X4_F32 (K-loop of 64).
// ---------------------------------------------------------------------------
__global__ __launch_bounds__(512) void proj_ln(const float* __restrict__ objs,
                                               const float* __restrict__ W,
                                               const float* __restrict__ bias,
                                               const float* __restrict__ gamma,
                                               const float* __restrict__ beta,
                                               float*       __restrict__ out) {
    __shared__ float tile[16 * D_];
    const int tid  = threadIdx.x;
    const int wave = tid >> 5;
    const int lane = tid & 31;
    const int half = lane >> 4;      // K-half selector per WMMA layout
    const int l    = lane & 15;      // row (A) / col (B,C,D) within tile
    const int rowbase = blockIdx.x * 16;
    const int colbase = wave * 16;

    v8f c = {};

#ifdef HAVE_F32_WMMA
    // A 16x4 f32: VGPR0 = K {0 | 2}, VGPR1 = K {1 | 3} by lane half.
    // B 4x16 f32: symmetric; B[k][n] = W[colbase+n][k]  (proj = objs * W^T).
    for (int k0 = 0; k0 < D_; k0 += 4) {
        const int k = k0 + 2 * half;
        v2f a, w;
        const float* ap = objs + (size_t)(rowbase + l) * D_ + k;
        a.x = ap[0]; a.y = ap[1];
        const float* wp = W + (size_t)(colbase + l) * D_ + k;
        w.x = wp[0]; w.y = wp[1];
        c = __builtin_amdgcn_wmma_f32_16x16x4_f32(false, a, false, w,
                                                  (short)0, c, false, false);
    }
#else
    // Fallback: f16 16x16x32 with on-the-fly convert (probe-confirmed builtin).
    for (int k0 = 0; k0 < D_; k0 += 32) {
        v16h a, w;
        const float* ap = objs + (size_t)(rowbase + l) * D_;
        const float* wp = W + (size_t)(colbase + l) * D_;
        for (int i = 0; i < 8; ++i) {
            const int kk = k0 + ((i < 4) ? (2 * i + 8 * half)
                                         : (16 + 2 * (i - 4) + 8 * half));
            a[2 * i]     = (_Float16)ap[kk];
            a[2 * i + 1] = (_Float16)ap[kk + 1];
            w[2 * i]     = (_Float16)wp[kk];
            w[2 * i + 1] = (_Float16)wp[kk + 1];
        }
        c = __builtin_amdgcn_wmma_f32_16x16x32_f16(false, a, false, w,
                                                   (short)0, c, false, false);
    }
#endif

    // C/D layout: VGPR j -> (M = j + 8*half, N = l). Add bias, stage strip.
    for (int j = 0; j < 8; ++j) {
        const int m = j + half * 8;
        const int n = colbase + l;
        tile[m * D_ + n] = c[j] + bias[n];
    }
    __syncthreads();

    // LayerNorm: wave w normalizes row w of the strip (256 elems, 8 per lane).
    const float* row = tile + wave * D_;
    float v[8], s = 0.0f, s2 = 0.0f;
    for (int j = 0; j < 8; ++j) {
        v[j] = row[lane * 8 + j];
        s  += v[j];
        s2 += v[j] * v[j];
    }
    for (int off = 16; off > 0; off >>= 1) {
        s  += __shfl_xor(s,  off, 32);
        s2 += __shfl_xor(s2, off, 32);
    }
    const float mean = s * (1.0f / D_);
    const float var  = s2 * (1.0f / D_) - mean * mean;
    const float rstd = rsqrtf(var + 1e-5f);
    float* orow = out + (size_t)(rowbase + wave) * D_;
    for (int j = 0; j < 8; ++j) {
        const int idx = lane * 8 + j;
        orow[idx] = (v[j] - mean) * rstd * gamma[idx] + beta[idx];
    }
}

// ---------------------------------------------------------------------------
extern "C" void kernel_launch(void* const* d_in, const int* in_sizes, int n_in,
                              void* d_out, int out_size, void* d_ws, size_t ws_size,
                              hipStream_t stream) {
    const float* features = (const float*)d_in[0];
    const int*   seg      = (const int*)  d_in[1];
    const float* W        = (const float*)d_in[2];
    const float* b        = (const float*)d_in[3];
    const float* gamma    = (const float*)d_in[4];
    const float* beta     = (const float*)d_in[5];
    float* ws  = (float*)d_ws;
    float* out = (float*)d_out;

    zero_ws  <<<512, 256, 0, stream>>>(ws);
    seg_accum<<<B_ * CHUNKS, 256, 0, stream>>>(features, seg, ws);
    top_objs <<<B_, 256, 0, stream>>>(ws);
    proj_ln  <<<(B_ * M_) / 16, 512, 0, stream>>>(ws + WS_OBJS, W, b, gamma, beta, out);
}
